// PSANetModule_55490977464693
// MI455X (gfx1250) — matrix operand order repaired and checked by reading, not compiled, not run on hardware
//
#include <hip/hip_runtime.h>

// ---------------------------------------------------------------------------
// PSANet forward for MI455X (gfx1250, wave32, WMMA 16x16x32 bf16)
// Memory-bound (~1 GB traffic @ 23.3 TB/s >> 0.7 TFLOP of bf16 WMMA), so all
// tensors are stored bf16, NHWC activations give contiguous K-major rows for
// the B operand, and every conv/einsum is one tiled WMMA GEMM kernel.
// ---------------------------------------------------------------------------

typedef __attribute__((ext_vector_type(16))) __bf16         v16bf;
typedef __attribute__((ext_vector_type(8)))  float          v8f;
typedef __attribute__((ext_vector_type(8)))  unsigned short u16x8;

#define BN_INV 0.9999949932f   // 1/sqrt(1+1e-5)

__device__ __forceinline__ unsigned short f2bf(float f) {
  union { float f; unsigned u; } v; v.f = f;
  return (unsigned short)((v.u + 0x7FFFu + ((v.u >> 16) & 1u)) >> 16); // RNE
}
__device__ __forceinline__ float bf2f(unsigned short h) {
  union { unsigned u; float f; } v; v.u = ((unsigned)h) << 16;
  return v.f;
}
__device__ __forceinline__ int imin(int a, int b) { return a < b ? a : b; }
__device__ __forceinline__ int imax(int a, int b) { return a > b ? a : b; }

// ---------------------------------------------------------------------------
// Generic bf16 GEMM:  D[m,n] = sum_k A[m,k] * Bt[n,k]   (Bt = B^T, row major)
// Epilogue: optional bias[m], optional BN+ReLU (g,bt), output either
//   - bf16 "NHWC": out[n*ldo + m]   (activation layout for the next GEMM)
//   - fp32 NCHW final: out[((n/hw)*M + m)*hw + n%hw]
// Block tile 128x64, 8 waves, each wave 2x2 WMMA 16x16 tiles, K step 32.
// Interior blocks take a uniform-branch fast path (pure b128 staging);
// edge blocks use clamped-address loads + select (no divergent branches).
// ---------------------------------------------------------------------------
#define TBM 128
#define TBN 64
#define TBK 32

__global__ __launch_bounds__(256)
void k_gemm_bf16(const unsigned short* __restrict__ A,
                 const unsigned short* __restrict__ Bt,
                 const float* __restrict__ bias,
                 const float* __restrict__ bng,
                 const float* __restrict__ bnb,
                 unsigned short* __restrict__ outBf,
                 float* __restrict__ outF32,
                 int M, int N, int K,
                 int lda, int ldb, int ldo,
                 long long strideA, long long strideB, long long strideO,
                 int hw)
{
  __shared__ __align__(16) unsigned short As[TBM * TBK];
  __shared__ __align__(16) unsigned short Bs[TBN * TBK];

  const int tid = threadIdx.x;
  const int m0  = blockIdx.y * TBM;
  const int n0  = blockIdx.x * TBN;

  A  += (long long)blockIdx.z * strideA;
  Bt += (long long)blockIdx.z * strideB;

  const int wave = tid >> 5;
  const int lane = tid & 31;
  const int wm   = (wave & 3) * 32;   // wave tile origin in M (4 waves)
  const int wn   = (wave >> 2) * 32;  // wave tile origin in N (2 waves)
  const int lr   = lane & 15;
  const int hi   = lane >> 4;         // half-wave select

  // Per-thread staging slots (fixed across K-steps)
  const int aRow0 = tid >> 2;               // chunk s=0: rows 0..63
  const int aRow1 = (tid + 256) >> 2;       // chunk s=1: rows 64..127
  const int aKc   = (tid & 3) * 8;
  const int bRow  = tid >> 2;               // rows 0..63
  const int bKc   = (tid & 3) * 8;

  const bool interior = (m0 + TBM <= M) && (n0 + TBN <= N);

  v8f zero = {0.f, 0.f, 0.f, 0.f, 0.f, 0.f, 0.f, 0.f};
  v8f acc[2][2];
  acc[0][0] = zero; acc[0][1] = zero; acc[1][0] = zero; acc[1][1] = zero;

  union Frag { u16x8 h[2]; v16bf v; };

  // Guarded staging: clamped-address loads + select-zero (branch-free).
  auto stage_guarded = [&](int k0) {
    #pragma unroll
    for (int s = 0; s < 2; ++s) {
      int row = s ? aRow1 : aRow0;
      int gm  = m0 + row;
      int gmc = imin(gm, M - 1);
      #pragma unroll
      for (int t = 0; t < 8; ++t) {
        int gk  = k0 + aKc + t;
        int gkc = imin(gk, K - 1);
        unsigned short v = A[(size_t)gmc * lda + gkc];
        As[row * TBK + aKc + t] = (gm < M && gk < K) ? v : (unsigned short)0;
      }
    }
    {
      int gn  = n0 + bRow;
      int gnc = imin(gn, N - 1);
      #pragma unroll
      for (int t = 0; t < 8; ++t) {
        int gk  = k0 + bKc + t;
        int gkc = imin(gk, K - 1);
        unsigned short v = Bt[(size_t)gnc * ldb + gkc];
        Bs[bRow * TBK + bKc + t] = (gn < N && gk < K) ? v : (unsigned short)0;
      }
    }
  };

  // Fast staging: pure 16B vector loads (requires interior && k0+TBK<=K).
  auto stage_fast = [&](int k0) {
    *(uint4*)&As[aRow0 * TBK + aKc] = *(const uint4*)(A + (size_t)(m0 + aRow0) * lda + k0 + aKc);
    *(uint4*)&As[aRow1 * TBK + aKc] = *(const uint4*)(A + (size_t)(m0 + aRow1) * lda + k0 + aKc);
    *(uint4*)&Bs[bRow  * TBK + bKc] = *(const uint4*)(Bt + (size_t)(n0 + bRow) * ldb + k0 + bKc);
    if (k0 + TBK < K) {  // prefetch next K tile -> global_prefetch_b8
      __builtin_prefetch(A + (size_t)(m0 + aRow0) * lda + k0 + TBK + aKc, 0, 1);
      __builtin_prefetch(Bt + (size_t)(n0 + bRow) * ldb + k0 + TBK + bKc, 0, 1);
    }
  };

  // Fragment loads per documented CDNA5 16-bit WMMA layouts + 2x2 WMMA.
  // A 16x32: lanes 0-15 row M=lr hold K 0-7 (v0-3) & 16-23 (v4-7);
  //          lanes 16-31 hold K 8-15 & 24-31.
// B 32x16: lanes 0-15 col N=lr hold K 0-7 & 16-23; lanes 16-31 hold K 8-15 & 24-31.
  auto compute = [&]() {
    Frag a[2], b[2];
    #pragma unroll
    for (int i = 0; i < 2; ++i) {
      const u16x8* ap = (const u16x8*)&As[(wm + i * 16 + lr) * TBK];
      a[i].h[0] = ap[hi];
      a[i].h[1] = ap[hi + 2];
    }
    #pragma unroll
    for (int j = 0; j < 2; ++j) {
      const u16x8* bp = (const u16x8*)&Bs[(wn + j * 16 + lr) * TBK];
      b[j].h[0] = bp[hi];
      b[j].h[1] = bp[hi + 2];
    }
    #pragma unroll
    for (int i = 0; i < 2; ++i)
      #pragma unroll
      for (int j = 0; j < 2; ++j)
        acc[i][j] = __builtin_amdgcn_wmma_f32_16x16x32_bf16(
            false, a[i].v, false, b[j].v, (short)0, acc[i][j], false, false);
  };

  const int kBody = K & ~(TBK - 1);   // K rounded down to TBK

  if (interior) {
    for (int k0 = 0; k0 < kBody; k0 += TBK) {   // hot loop: branch-free
      __syncthreads();
      stage_fast(k0);
      __syncthreads();
      compute();
    }
    if (kBody < K) {                            // K tail (e.g. K=900)
      __syncthreads();
      stage_guarded(kBody);
      __syncthreads();
      compute();
    }
  } else {
    for (int k0 = 0; k0 < K; k0 += TBK) {
      __syncthreads();
      stage_guarded(k0);
      __syncthreads();
      compute();
    }
  }

  // ---- epilogue: C/D layout: lane(lr,hi), VGPR r -> (m = r + 8*hi, n = lr) ----
  #pragma unroll
  for (int j = 0; j < 2; ++j) {
    int gn = n0 + wn + j * 16 + lr;
    if (gn >= N) continue;
    #pragma unroll
    for (int i = 0; i < 2; ++i) {
      #pragma unroll
      for (int r = 0; r < 8; ++r) {
        int gm = m0 + wm + i * 16 + hi * 8 + r;
        if (gm >= M) continue;
        float v = acc[i][j][r];
        if (bias) v += bias[gm];
        if (bng)  v = fmaxf(bng[gm] * (v * BN_INV) + bnb[gm], 0.f);
        if (outBf) {
          outBf[(long long)blockIdx.z * strideO + (size_t)gn * ldo + gm] = f2bf(v);
        } else {
          outF32[((size_t)(gn / hw) * (size_t)M + (size_t)gm) * (size_t)hw + (gn % hw)] = v;
        }
      }
    }
  }
}

// ---------------------------------------------------------------------------
// Helpers / layout kernels
// ---------------------------------------------------------------------------
__global__ void k_f32_to_bf16(const float* __restrict__ in,
                              unsigned short* __restrict__ out, int n) {
  int i = blockIdx.x * blockDim.x + threadIdx.x;
  if (i < n) out[i] = f2bf(in[i]);
}

// x fp32 NCHW [8,2048,60,60] -> cat bf16 NHWC [8*3600, 4096] channels [0,2048)
__global__ void k_cvt_x(const float* __restrict__ x,
                        unsigned short* __restrict__ cat) {
  int i = blockIdx.x * blockDim.x + threadIdx.x;
  if (i >= 8 * 3600 * 2048) return;
  int c = i & 2047;
  int p = (i >> 11) % 3600;
  int b = i / (3600 * 2048);
  cat[((size_t)(b * 3600 + p)) * 4096 + c] = f2bf(x[((size_t)b * 2048 + c) * 3600 + p]);
}

// AvgPool2d(3,2,1) count_include_pad: NHWC bf16 [8,60,60,512] ->
//   NHWC bf16 [8,900,512] and NCHW bf16 [8,512,912] (ld padded to 912)
__global__ void k_pool(const unsigned short* __restrict__ in,
                       unsigned short* __restrict__ out_nhwc,
                       unsigned short* __restrict__ out_nchw) {
  const int C = 512;
  int i = blockIdx.x * blockDim.x + threadIdx.x;
  if (i >= 8 * 900 * C) return;
  int c = i & (C - 1);
  int t = i >> 9;
  int ow = t % 30; t /= 30;
  int oh = t % 30;
  int b  = t / 30;
  float s = 0.f;
  for (int dy = -1; dy <= 1; ++dy) {
    int ih = 2 * oh + dy;
    if (ih < 0 || ih >= 60) continue;
    for (int dx = -1; dx <= 1; ++dx) {
      int iw = 2 * ow + dx;
      if (iw < 0 || iw >= 60) continue;
      s += bf2f(in[(((size_t)b * 60 + ih) * 60 + iw) * C + c]);
    }
  }
  unsigned short r = f2bf(s * (1.f / 9.f));
  int p = oh * 30 + ow;
  out_nhwc[((size_t)b * 900 + p) * C + c] = r;
  out_nchw[((size_t)b * C + c) * 912 + p] = r;
}

// Build attention matrix (bf16):
//   collect:    G[b,i,j] = oc[b, i, idx(i,j)]
//   distribute: G[b,i,j] = oc[b, j, idx(j,i)]
// oc: [8,900,3488] (ld padded); G: [8,900,912] (ld padded)
__global__ void k_gather(const unsigned short* __restrict__ oc,
                         unsigned short* __restrict__ G, int distribute) {
  int idx = blockIdx.x * blockDim.x + threadIdx.x;
  if (idx >= 8 * 900 * 900) return;
  int j = idx % 900;
  int t = idx / 900;
  int i = t % 900;
  int b = t / 900;
  int hi_ = i / 30, wi_ = i % 30, hj_ = j / 30, wj_ = j % 30;
  int ih, iw, row;
  if (distribute) { ih = 29 - hj_ + hi_; iw = 29 - wj_ + wi_; row = j; }
  else            { ih = 29 - hi_ + hj_; iw = 29 - wi_ + wj_; row = i; }
  G[((size_t)b * 900 + i) * 912 + j] =
      oc[((size_t)b * 900 + row) * 3488 + ih * 59 + iw];
}

// Bilinear 2x upsample (half-pixel, edge clamp): psa NHWC bf16 [8,30,30,2048]
// -> cat NHWC bf16 [8*3600,4096] channels [2048,4096)
__global__ void k_upsample(const unsigned short* __restrict__ psa,
                           unsigned short* __restrict__ cat) {
  int i = blockIdx.x * blockDim.x + threadIdx.x;
  if (i >= 8 * 3600 * 2048) return;
  int c = i & 2047;
  int p = (i >> 11) % 3600;
  int b = i / (3600 * 2048);
  int oh = p / 60, ow = p % 60;
  float sh = 0.5f * oh - 0.25f;
  float sw = 0.5f * ow - 0.25f;
  int h0 = (int)floorf(sh); float fh = sh - (float)h0;
  int w0 = (int)floorf(sw); float fw = sw - (float)w0;
  int h0c = imin(imax(h0, 0), 29), h1c = imin(imax(h0 + 1, 0), 29);
  int w0c = imin(imax(w0, 0), 29), w1c = imin(imax(w0 + 1, 0), 29);
  const size_t bb = (size_t)b * 900;
  float v00 = bf2f(psa[(bb + h0c * 30 + w0c) * 2048 + c]);
  float v01 = bf2f(psa[(bb + h0c * 30 + w1c) * 2048 + c]);
  float v10 = bf2f(psa[(bb + h1c * 30 + w0c) * 2048 + c]);
  float v11 = bf2f(psa[(bb + h1c * 30 + w1c) * 2048 + c]);
  float v = (1.f - fh) * ((1.f - fw) * v00 + fw * v01) +
            fh * ((1.f - fw) * v10 + fw * v11);
  cat[((size_t)(b * 3600 + p)) * 4096 + 2048 + c] = f2bf(v);
}

// ---------------------------------------------------------------------------
extern "C" void kernel_launch(void* const* d_in, const int* in_sizes, int n_in,
                              void* d_out, int out_size, void* d_ws, size_t ws_size,
                              hipStream_t stream) {
  (void)in_sizes; (void)n_in; (void)out_size; (void)ws_size;

  const float* x       = (const float*)d_in[0];
  const float* c1_w    = (const float*)d_in[1];
  const float* c1_b    = (const float*)d_in[2];
  const float* c1_g    = (const float*)d_in[3];
  const float* c1_bt   = (const float*)d_in[4];
  const float* d1_w    = (const float*)d_in[5];
  const float* d1_b    = (const float*)d_in[6];
  const float* d1_g    = (const float*)d_in[7];
  const float* d1_bt   = (const float*)d_in[8];
  const float* c2a_w   = (const float*)d_in[9];
  const float* c2a_b   = (const float*)d_in[10];
  const float* c2a_g   = (const float*)d_in[11];
  const float* c2a_bt  = (const float*)d_in[12];
  const float* d2a_w   = (const float*)d_in[13];
  const float* d2a_b   = (const float*)d_in[14];
  const float* d2a_g   = (const float*)d_in[15];
  const float* d2a_bt  = (const float*)d_in[16];
  const float* c2b_w   = (const float*)d_in[17];
  const float* c2b_b   = (const float*)d_in[18];
  const float* d2b_w   = (const float*)d_in[19];
  const float* d2b_b   = (const float*)d_in[20];
  const float* en_w    = (const float*)d_in[21];
  const float* en_b    = (const float*)d_in[22];
  const float* en_g    = (const float*)d_in[23];
  const float* en_bt   = (const float*)d_in[24];
  const float* last_w  = (const float*)d_in[25];
  const float* last_b  = (const float*)d_in[26];

  char* ws = (char*)d_ws;
  size_t off = 0;
  auto take = [&](size_t bytes) -> void* {
    void* p = ws + off;
    off = (off + bytes + 255) & ~(size_t)255;
    return p;
  };

  // bf16 buffers (all leading dims multiples of 16 elements for b128 loads)
  unsigned short* cat     = (unsigned short*)take((size_t)28800 * 4096 * 2); // x || up
  unsigned short* xc_full = (unsigned short*)take((size_t)28800 * 512 * 2);
  unsigned short* xd_full = (unsigned short*)take((size_t)28800 * 512 * 2);
  unsigned short* xc_nhwc = (unsigned short*)take((size_t)7200 * 512 * 2);
  unsigned short* xc_nchw = (unsigned short*)take((size_t)8 * 512 * 912 * 2);
  unsigned short* xd_nhwc = (unsigned short*)take((size_t)7200 * 512 * 2);
  unsigned short* xd_nchw = (unsigned short*)take((size_t)8 * 512 * 912 * 2);
  unsigned short* hbuf    = (unsigned short*)take((size_t)7200 * 512 * 2);   // reused c/d
  unsigned short* ocbuf   = (unsigned short*)take((size_t)7200 * 3488 * 2);  // reused c/d
  unsigned short* Gbuf    = (unsigned short*)take((size_t)8 * 900 * 912 * 2);// reused c/d
  unsigned short* en_in   = (unsigned short*)take((size_t)7200 * 1024 * 2);  // col || dis
  unsigned short* psa     = (unsigned short*)take((size_t)7200 * 2048 * 2);
  unsigned short* w_c1    = (unsigned short*)take((size_t)512 * 2048 * 2);
  unsigned short* w_d1    = (unsigned short*)take((size_t)512 * 2048 * 2);
  unsigned short* w_c2a   = (unsigned short*)take((size_t)512 * 512 * 2);
  unsigned short* w_d2a   = (unsigned short*)take((size_t)512 * 512 * 2);
  unsigned short* w_c2b   = (unsigned short*)take((size_t)3481 * 512 * 2);
  unsigned short* w_d2b   = (unsigned short*)take((size_t)3481 * 512 * 2);
  unsigned short* w_en    = (unsigned short*)take((size_t)2048 * 1024 * 2);
  unsigned short* w_last  = (unsigned short*)take((size_t)2048 * 4096 * 2);

  auto cdiv = [](int a, int b) { return (a + b - 1) / b; };
  dim3 blk(256);

  // --- weights fp32 -> bf16 ---
  k_f32_to_bf16<<<cdiv(512*2048, 256), blk, 0, stream>>>(c1_w,  w_c1,  512*2048);
  k_f32_to_bf16<<<cdiv(512*2048, 256), blk, 0, stream>>>(d1_w,  w_d1,  512*2048);
  k_f32_to_bf16<<<cdiv(512*512,  256), blk, 0, stream>>>(c2a_w, w_c2a, 512*512);
  k_f32_to_bf16<<<cdiv(512*512,  256), blk, 0, stream>>>(d2a_w, w_d2a, 512*512);
  k_f32_to_bf16<<<cdiv(3481*512, 256), blk, 0, stream>>>(c2b_w, w_c2b, 3481*512);
  k_f32_to_bf16<<<cdiv(3481*512, 256), blk, 0, stream>>>(d2b_w, w_d2b, 3481*512);
  k_f32_to_bf16<<<cdiv(2048*1024,256), blk, 0, stream>>>(en_w,  w_en,  2048*1024);
  k_f32_to_bf16<<<cdiv(2048*4096,256), blk, 0, stream>>>(last_w,w_last,2048*4096);

  // --- x -> cat[:, 0:2048] (NHWC bf16) ---
  k_cvt_x<<<cdiv(8*3600*2048, 256), blk, 0, stream>>>(x, cat);

  auto gemm = [&](const unsigned short* A, const unsigned short* Bt,
                  const float* bias, const float* g, const float* bt,
                  unsigned short* oB, float* oF,
                  int M, int N, int K, int lda, int ldb, int ldo,
                  long long sA, long long sB, long long sO, int nb, int hw) {
    dim3 grid(cdiv(N, TBN), cdiv(M, TBM), nb);
    k_gemm_bf16<<<grid, blk, 0, stream>>>(A, Bt, bias, g, bt, oB, oF,
                                          M, N, K, lda, ldb, ldo, sA, sB, sO, hw);
  };

  // --- stage 1: c1/d1 + BN + ReLU ---
  gemm(w_c1, cat, c1_b, c1_g, c1_bt, xc_full, nullptr,
       512, 28800, 2048, 2048, 4096, 512, 0, 0, 0, 1, 0);
  gemm(w_d1, cat, d1_b, d1_g, d1_bt, xd_full, nullptr,
       512, 28800, 2048, 2048, 4096, 512, 0, 0, 0, 1, 0);
  k_pool<<<cdiv(8*900*512, 256), blk, 0, stream>>>(xc_full, xc_nhwc, xc_nchw);
  k_pool<<<cdiv(8*900*512, 256), blk, 0, stream>>>(xd_full, xd_nhwc, xd_nchw);

  // --- collect path ---
  gemm(w_c2a, xc_nhwc, c2a_b, c2a_g, c2a_bt, hbuf, nullptr,
       512, 7200, 512, 512, 512, 512, 0, 0, 0, 1, 0);
  gemm(w_c2b, hbuf, c2b_b, nullptr, nullptr, ocbuf, nullptr,
       3481, 7200, 512, 512, 512, 3488, 0, 0, 0, 1, 0);
  k_gather<<<cdiv(8*900*900, 256), blk, 0, stream>>>(ocbuf, Gbuf, 0);
  // col[i,c] = sum_j Xc_nchw[c,j] * G_c[i,j]   (batched, 8)
  gemm(xc_nchw, Gbuf, nullptr, nullptr, nullptr, en_in, nullptr,
       512, 900, 900, 912, 912, 1024,
       (long long)512 * 912, (long long)900 * 912, (long long)900 * 1024, 8, 0);

  // --- distribute path (reuses hbuf/ocbuf/Gbuf sequentially) ---
  gemm(w_d2a, xd_nhwc, d2a_b, d2a_g, d2a_bt, hbuf, nullptr,
       512, 7200, 512, 512, 512, 512, 0, 0, 0, 1, 0);
  gemm(w_d2b, hbuf, d2b_b, nullptr, nullptr, ocbuf, nullptr,
       3481, 7200, 512, 512, 512, 3488, 0, 0, 0, 1, 0);
  k_gather<<<cdiv(8*900*900, 256), blk, 0, stream>>>(ocbuf, Gbuf, 1);
  gemm(xd_nchw, Gbuf, nullptr, nullptr, nullptr, en_in + 512, nullptr,
       512, 900, 900, 912, 912, 1024,
       (long long)512 * 912, (long long)900 * 912, (long long)900 * 1024, 8, 0);

  // --- encode: en conv on concat(col, dis) + BN + ReLU ---
  gemm(w_en, en_in, en_b, en_g, en_bt, psa, nullptr,
       2048, 7200, 1024, 1024, 1024, 2048, 0, 0, 0, 1, 0);

  // --- bilinear upsample -> cat[:, 2048:4096] ---
  k_upsample<<<cdiv(8*3600*2048, 256), blk, 0, stream>>>(psa, cat);

  // --- final conv on concat(x, up) -> fp32 NCHW d_out ---
  gemm(w_last, cat, last_b, nullptr, nullptr, nullptr, (float*)d_out,
       2048, 28800, 4096, 4096, 4096, 0, 0, 0, 0, 1, 3600);
}
